// BaselineMoELayer_71425306132871
// MI455X (gfx1250) — matrix operand order repaired and checked by reading, not compiled, not run on hardware
//
#include <hip/hip_runtime.h>
#include <hip/hip_bf16.h>
#include <stdint.h>

// ---------------- constants ----------------
#define NTOK 8192      // B*S
#define DDIM 1024
#define HDIM 4096
#define NEXP 8
#define XS_STRIDE 1040   // 1024 + 16 halves pad (bank spread)
#define HS_STRIDE 4112   // 4096 + 16 halves pad

typedef __attribute__((ext_vector_type(16))) _Float16 v16h;
typedef __attribute__((ext_vector_type(8)))  _Float16 v8h;
typedef __attribute__((ext_vector_type(8)))  float    v8f;

// ---------------- WMMA fragment helpers (wave32, 16x16x32 f16) ----------------
// A (16x32 f16): lane<16 -> row M=lane, K [0..15]; lane>=16 -> M=lane-16, K [16..31]
__device__ __forceinline__ v16h load_a_frag_lds(const _Float16* base, int stride, int k0, int lane) {
    int row = lane & 15;
    int kh  = (lane < 16) ? 0 : 16;
    const _Float16* p = base + row * stride + k0 + kh;
    v8h lo = *(const v8h*)p;          // ds_load_b128
    v8h hi = *(const v8h*)(p + 8);    // ds_load_b128
    v16h a;
#pragma unroll
    for (int i = 0; i < 8; ++i) { a[i] = lo[i]; a[8 + i] = hi[i]; }
    return a;
}

// ---------------- init: zero per-expert counters ----------------
__global__ void moe_init_kernel(int* __restrict__ counts) {
    if (threadIdx.x < NEXP) counts[threadIdx.x] = 0;
}

// ---------------- f32 -> f16 convert ----------------
__global__ void moe_cvt_x_kernel(const float* __restrict__ in, _Float16* __restrict__ out, int n) {
    int i = blockIdx.x * blockDim.x + threadIdx.x;
    if (i < n) out[i] = (_Float16)in[i];
}

// in: [E][rows][cols] f32  ->  out: [E][cols][rows] f16
__global__ void moe_transpose_w_kernel(const float* __restrict__ in, _Float16* __restrict__ out,
                                       int rows, int cols) {
    size_t i   = (size_t)blockIdx.x * blockDim.x + threadIdx.x;
    size_t per = (size_t)rows * cols;
    size_t e   = i / per;
    size_t rem = i - e * per;
    size_t r   = rem / cols;
    size_t c   = rem - r * cols;
    out[e * per + c * rows + r] = (_Float16)in[i];
}

// ---------------- router: logits, softmax, top-2, scatter lists ----------------
__global__ void moe_router_kernel(const float* __restrict__ x,
                                  const float* __restrict__ rW,   // [D, E]
                                  const float* __restrict__ rb,   // [E]
                                  float* __restrict__ gating,     // [N, E]  (d_out)
                                  int*   __restrict__ topidx,     // [N, 2]  (d_out)
                                  float* __restrict__ topw,       // [N, 2]  (ws)
                                  int*   __restrict__ lists,      // [E, N]
                                  int*   __restrict__ counts) {   // [E]
    const int wave = threadIdx.x >> 5;
    const int lane = threadIdx.x & 31;
    const int t = blockIdx.x * 8 + wave;

    float acc[NEXP];
#pragma unroll
    for (int e = 0; e < NEXP; ++e) acc[e] = 0.f;

    const float* xr = x + (size_t)t * DDIM;
    for (int d = lane; d < DDIM; d += 32) {
        float xv = xr[d];
        const float4* wr = (const float4*)(rW + d * NEXP);
        float4 wa = wr[0], wb = wr[1];
        acc[0] += xv * wa.x; acc[1] += xv * wa.y; acc[2] += xv * wa.z; acc[3] += xv * wa.w;
        acc[4] += xv * wb.x; acc[5] += xv * wb.y; acc[6] += xv * wb.z; acc[7] += xv * wb.w;
    }
#pragma unroll
    for (int e = 0; e < NEXP; ++e)
        for (int off = 16; off > 0; off >>= 1)
            acc[e] += __shfl_xor(acc[e], off, 32);

    if (lane == 0) {
        float l[NEXP], ex[NEXP];
        float mx = -1e30f;
#pragma unroll
        for (int e = 0; e < NEXP; ++e) { l[e] = acc[e] + rb[e]; mx = fmaxf(mx, l[e]); }
        float s = 0.f;
#pragma unroll
        for (int e = 0; e < NEXP; ++e) { ex[e] = __expf(l[e] - mx); s += ex[e]; }
        float inv = 1.f / s;
#pragma unroll
        for (int e = 0; e < NEXP; ++e) gating[(size_t)t * NEXP + e] = ex[e] * inv;

        int i0 = 0;
#pragma unroll
        for (int e = 1; e < NEXP; ++e) if (l[e] > l[i0]) i0 = e;     // ties -> lowest idx
        int i1 = (i0 == 0) ? 1 : 0;
#pragma unroll
        for (int e = 0; e < NEXP; ++e) if (e != i0 && l[e] > l[i1]) i1 = e;

        float p0 = 1.f / (1.f + __expf(l[i1] - l[i0]));              // renorm over top-2
        float p1 = 1.f - p0;
        topidx[t * 2]     = i0;
        topidx[t * 2 + 1] = i1;
        topw[t * 2]       = p0;
        topw[t * 2 + 1]   = p1;

        int pos0 = atomicAdd(&counts[i0], 1);
        lists[i0 * NTOK + pos0] = t * 2;       // (token<<1)|slot
        int pos1 = atomicAdd(&counts[i1], 1);
        lists[i1 * NTOK + pos1] = t * 2 + 1;
    }
}

// ---------------- fused expert FFN: 16-token tile, WMMA both GEMMs ----------------
__global__ void moe_ffn_kernel(const _Float16* __restrict__ xf16,   // [N, D]
                               const _Float16* __restrict__ W1t,    // [E][H][D]  (transposed)
                               const _Float16* __restrict__ W2t,    // [E][D][H]  (transposed)
                               const float* __restrict__ b1,        // [E, H]
                               const float* __restrict__ b2,        // [E, D]
                               const int* __restrict__ lists,       // [E, N]
                               const int* __restrict__ counts,      // [E]
                               float* __restrict__ ybuf) {          // [N*2, D]
    extern __shared__ char smem[];
    _Float16* xs   = (_Float16*)smem;                    // 16 x XS_STRIDE halves
    _Float16* hs   = xs + 16 * XS_STRIDE;                // 16 x HS_STRIDE halves
    int*      rowi = (int*)(hs + 16 * HS_STRIDE);        // 16 ints

    const int e   = blockIdx.y;
    const int cnt = counts[e];
    const int m0  = blockIdx.x * 16;
    if (m0 >= cnt) return;

    const int tid  = threadIdx.x;
    const int lane = tid & 31;
    const int wave = tid >> 5;
    const int col  = lane & 15;
    const int mh   = (lane < 16) ? 0 : 8;
    const int khB  = (lane < 16) ? 0 : 16;   // B-fragment K half offset

    if (tid < 16) {
        int m = m0 + tid;
        rowi[tid] = (m < cnt) ? lists[e * NTOK + m] : -1;
    }
    __syncthreads();

    // gather 16 x-rows (f16) into LDS; padding rows zeroed
    {
        int r = tid >> 4;   // 0..15
        int c = tid & 15;   // 16 threads per row, 8 uint4 each (128 uint4/row)
        int aid = rowi[r];
        uint4* dst = (uint4*)(xs + r * XS_STRIDE);
        if (aid >= 0) {
            const uint4* src = (const uint4*)(xf16 + (size_t)(aid >> 1) * DDIM);
#pragma unroll
            for (int i = 0; i < 8; ++i) dst[c * 8 + i] = src[c * 8 + i];
        } else {
            uint4 z = {0, 0, 0, 0};
#pragma unroll
            for (int i = 0; i < 8; ++i) dst[c * 8 + i] = z;
        }
    }
    __syncthreads();

    // ---- Phase 1: h = relu(x @ W1 + b1), wave owns 32 N-tiles (512 cols of H) ----
    const _Float16* w1e = W1t + (size_t)e * HDIM * DDIM;
    for (int i = 0; i < 32; ++i) {
        const int n0 = (wave * 32 + i) * 16;
        const _Float16* pb = w1e + (size_t)(n0 + col) * DDIM + khB;
        v8f c = {};
        for (int k0 = 0; k0 < DDIM; k0 += 32) {
            v16h a = load_a_frag_lds(xs, XS_STRIDE, k0, lane);
            v16h b = *(const v16h*)(pb + k0);                 // 2x global_load_b128
            __builtin_prefetch((const void*)(pb + k0 + 64), 0, 0);
            c = __builtin_amdgcn_wmma_f32_16x16x32_f16(false, a, false, b,
                                                       (short)0, c, false, false);
        }
        float bias = b1[e * HDIM + n0 + col];
#pragma unroll
        for (int r = 0; r < 8; ++r) {
            float v = c[r] + bias;
            v = v > 0.f ? v : 0.f;
            hs[(mh + r) * HS_STRIDE + n0 + col] = (_Float16)v;
        }
    }
    __syncthreads();

    // ---- Phase 2: y = h @ W2 + b2, wave owns 8 N-tiles (128 cols of D) ----
    const _Float16* w2e = W2t + (size_t)e * DDIM * HDIM;
    for (int i = 0; i < 8; ++i) {
        const int n0 = (wave * 8 + i) * 16;
        const _Float16* pb = w2e + (size_t)(n0 + col) * HDIM + khB;
        v8f c = {};
        for (int k0 = 0; k0 < HDIM; k0 += 32) {
            v16h a = load_a_frag_lds(hs, HS_STRIDE, k0, lane);
            v16h b = *(const v16h*)(pb + k0);
            __builtin_prefetch((const void*)(pb + k0 + 64), 0, 0);
            c = __builtin_amdgcn_wmma_f32_16x16x32_f16(false, a, false, b,
                                                       (short)0, c, false, false);
        }
        float bias = b2[e * DDIM + n0 + col];
#pragma unroll
        for (int r = 0; r < 8; ++r) {
            int aid = rowi[mh + r];
            if (aid >= 0) ybuf[(size_t)aid * DDIM + n0 + col] = c[r] + bias;
        }
    }
}

// ---------------- combine: out[t] = w0*y[t,0] + w1*y[t,1] (deterministic) ----------------
__global__ void moe_combine_kernel(const float* __restrict__ ybuf,
                                   const float* __restrict__ topw,
                                   float* __restrict__ out) {
    size_t i = (size_t)blockIdx.x * blockDim.x + threadIdx.x;   // over N*D
    size_t t = i >> 10;
    size_t d = i & 1023;
    float w0 = topw[t * 2], w1 = topw[t * 2 + 1];
    out[i] = w0 * ybuf[(t * 2) * DDIM + d] + w1 * ybuf[(t * 2 + 1) * DDIM + d];
}

// ---------------- host launcher ----------------
extern "C" void kernel_launch(void* const* d_in, const int* in_sizes, int n_in,
                              void* d_out, int out_size, void* d_ws, size_t ws_size,
                              hipStream_t stream) {
    const float* x   = (const float*)d_in[0];
    const float* rW  = (const float*)d_in[1];
    const float* rb  = (const float*)d_in[2];
    const float* W1  = (const float*)d_in[3];
    const float* b1  = (const float*)d_in[4];
    const float* W2  = (const float*)d_in[5];
    const float* b2  = (const float*)d_in[6];

    float* out    = (float*)d_out;                       // [N, D]
    float* gating = out + (size_t)NTOK * DDIM;           // [N, E]
    int*   topidx = (int*)(gating + (size_t)NTOK * NEXP);// [N, 2]

    // workspace carve-up (256B aligned)
    char* ws = (char*)d_ws;
    size_t off = 0;
    auto carve = [&](size_t bytes) -> void* {
        void* p = ws + off;
        off = (off + bytes + 255) & ~(size_t)255;
        return p;
    };
    _Float16* xf16 = (_Float16*)carve((size_t)NTOK * DDIM * 2);
    _Float16* W1t  = (_Float16*)carve((size_t)NEXP * DDIM * HDIM * 2);
    _Float16* W2t  = (_Float16*)carve((size_t)NEXP * DDIM * HDIM * 2);
    float*    ybuf = (float*)   carve((size_t)NTOK * 2 * DDIM * 4);
    int*      lists = (int*)    carve((size_t)NEXP * NTOK * 4);
    float*    topw  = (float*)  carve((size_t)NTOK * 2 * 4);
    int*      counts = (int*)   carve(256);
    (void)ws_size; (void)in_sizes; (void)n_in; (void)out_size;

    moe_init_kernel<<<1, 32, 0, stream>>>(counts);

    moe_cvt_x_kernel<<<(NTOK * DDIM) / 256, 256, 0, stream>>>(x, xf16, NTOK * DDIM);

    const int wElems = NEXP * DDIM * HDIM;
    moe_transpose_w_kernel<<<wElems / 256, 256, 0, stream>>>(W1, W1t, DDIM, HDIM); // [E][D][H]->[E][H][D]
    moe_transpose_w_kernel<<<wElems / 256, 256, 0, stream>>>(W2, W2t, HDIM, DDIM); // [E][H][D]->[E][D][H]

    moe_router_kernel<<<NTOK / 8, 256, 0, stream>>>(x, rW, rb, gating, topidx, topw, lists, counts);

    const size_t smem = (size_t)(16 * XS_STRIDE + 16 * HS_STRIDE) * 2 + 64; // ~165 KB of 320 KB LDS
    dim3 ffnGrid(NTOK / 16, NEXP);
    moe_ffn_kernel<<<ffnGrid, 256, smem, stream>>>(xf16, W1t, W2t, b1, b2, lists, counts, ybuf);

    moe_combine_kernel<<<(NTOK * DDIM) / 256, 256, 0, stream>>>(ybuf, topw, out);
}
